// MultiHeadAttention_13941463843500
// MI455X (gfx1250) — matrix operand order repaired and checked by reading, not compiled, not run on hardware
//
#include <hip/hip_runtime.h>
#include <hip/hip_bf16.h>

// ---------- types ----------
typedef __attribute__((ext_vector_type(16))) __bf16 v16bf;
typedef __attribute__((ext_vector_type(8)))  float  v8f;

union FragBF {
    v16bf v;
    uint4 q[2];
    unsigned short us[16];
};

// f32 -> bf16 (round-half-up), 2 VALU
__device__ inline unsigned short f2bf(float f) {
    return (unsigned short)((__float_as_uint(f) + 0x8000u) >> 16);
}

// pack two f32 -> bf16x2 in one dword
#if __has_builtin(__builtin_amdgcn_cvt_pk_bf16_f32)
typedef __attribute__((ext_vector_type(2))) __bf16 v2bf;
__device__ inline unsigned pk2bf(float a, float b) {
    union { v2bf v; unsigned u; } c;
    c.v = __builtin_amdgcn_cvt_pk_bf16_f32(a, b);
    return c.u;
}
#else
// v_perm_b32: D.byte[i] = sel.byte[i] of {S0(bytes 4-7), S1(bytes 0-3)}
// low half = a[31:16], high half = b[31:16]
__device__ inline unsigned pk2bf(float a, float b) {
    unsigned ua = __float_as_uint(a) + 0x8000u;
    unsigned ub = __float_as_uint(b) + 0x8000u;
    return __builtin_amdgcn_perm(ub, ua, 0x07060302u);
}
#endif

// pack 8 floats -> 8 bf16 in a uint4
__device__ inline uint4 cvt8_bf16(float4 a, float4 b) {
    uint4 r;
    r.x = pk2bf(a.x, a.y);
    r.y = pk2bf(a.z, a.w);
    r.z = pk2bf(b.x, b.y);
    r.w = pk2bf(b.z, b.w);
    return r;
}

__device__ inline v8f wmma_bf16(v16bf a, v16bf b, v8f c) {
    // D = A(16x32 bf16) * B(32x16 bf16) + C(16x16 f32)
    return __builtin_amdgcn_wmma_f32_16x16x32_bf16(
        false, a, false, b, (short)0, c, false, false);
}

// A fragment: 16x32 tile, element (m,k) at base[m*stride + k].
__device__ inline v16bf load_a_frag(const unsigned short* base, int stride, int lane) {
    int m  = lane & 15;
    int lh = lane >> 4;
    FragBF f;
    f.q[0] = *(const uint4*)(base + m * stride + lh * 8);
    f.q[1] = *(const uint4*)(base + m * stride + 16 + lh * 8);
    return f.v;
}

// B fragment: 32x16 tile stored column-per-row: base[n*stride + k] = B[k][n].
__device__ inline v16bf load_b_frag(const unsigned short* base, int stride, int lane) {
    int n  = lane & 15;
    int kh = lane >> 4;
    FragBF f;
    const unsigned short* p = base + n * stride + kh * 8;
    f.q[0] = *(const uint4*)(p);
    f.q[1] = *(const uint4*)(p + 16);
    return f.v;
}

// ---------- CDNA5 Tensor Data Mover ----------
#if __has_builtin(__builtin_amdgcn_tensor_load_to_lds)
#define HAVE_TDM 1
typedef __attribute__((ext_vector_type(4))) unsigned int u32x4;
typedef __attribute__((ext_vector_type(4))) int          i32x4;
typedef __attribute__((ext_vector_type(8))) int          i32x8;
#endif

// generic pointer -> LDS byte offset
__device__ inline unsigned lds_off_of(const void* p) {
    return (unsigned)(unsigned long long)(__attribute__((address_space(3))) const void*)p;
}

#ifdef HAVE_TDM
// DMA a 2D tile of bf16: tw elems per row, th rows, global row stride (elems).
// D# per CDNA5 ISA.
__device__ inline void tdm_load_2d(const void* gsrc, unsigned lds_off,
                                   int tw, int th, int stride) {
    unsigned long long ga = (unsigned long long)gsrc;
    u32x4 g0;
    g0[0] = 1u;                                                  // count=1, user D#
    g0[1] = lds_off;                                             // lds_addr
    g0[2] = (unsigned)ga;                                        // global_addr[31:0]
    g0[3] = (unsigned)((ga >> 32) & 0x01FFFFFFull) | (2u << 30); // addr[56:32] | type=2
    i32x8 g1;
    g1[0] = 1 << 16;                  // workgroup_mask=0, data_size=1 (2 bytes)
    g1[1] = (tw & 0xffff) << 16;      // tensor_dim0 (bits 79:48, low half)
    g1[2] = (th & 0xffff) << 16;      // tensor_dim0 hi=0 | tensor_dim1 low16
    g1[3] = (tw & 0xffff) << 16;      // tensor_dim1 hi=0 | tile_dim0
    g1[4] = th & 0xffff;              // tile_dim1, tile_dim2=0
    g1[5] = stride;                   // tensor_dim0_stride low32
    g1[6] = 0;
    g1[7] = 0;
    i32x4 z4 = {0, 0, 0, 0};
#if __clang_major__ >= 23
    i32x8 z8 = {0, 0, 0, 0, 0, 0, 0, 0};
    __builtin_amdgcn_tensor_load_to_lds(g0, g1, z4, z4, z8, 0);
#else
    __builtin_amdgcn_tensor_load_to_lds(g0, g1, z4, z4, 0);
#endif
}
#endif

__device__ inline void wait_tensorcnt0() {
#if __has_builtin(__builtin_amdgcn_s_wait_tensorcnt)
    __builtin_amdgcn_s_wait_tensorcnt(0);
#else
    asm volatile("s_wait_tensorcnt 0" ::: "memory");
#endif
}

// ---------------------------------------------------------------------------
// OUT = X[M,1024] @ W[1024,1024]^T + bias   (torch Linear semantics)
// mode 0: f32 row-major [M,1024]
// mode 1: bf16 in [B=2, H=16, S=2048, HD=64]   (Q, K)
// mode 2: bf16 in [B=2, H=16, HD=64, S=2048]   (V, pre-transposed per head)
// ---------------------------------------------------------------------------
__global__ __launch_bounds__(256)
void gemm_bias_kernel(const float* __restrict__ X, const float* __restrict__ W,
                      const float* __restrict__ bias, void* __restrict__ out, int mode)
{
    __shared__ __align__(16) unsigned short Atile[128 * 32];
    __shared__ __align__(16) unsigned short Btile[64 * 32];

    const int tid  = threadIdx.x;
    const int lane = tid & 31;
    const int wave = tid >> 5;
    const int n0   = blockIdx.x * 64;
    const int m0   = blockIdx.y * 128;

    v8f acc[4] = {};

    for (int k0 = 0; k0 < 1024; k0 += 32) {
        // stage A: 128 rows x 32 cols f32 -> bf16 (2 threads per row)
        {
            int row = tid >> 1;
            int c0  = (tid & 1) * 16;
            const float* base = X + (size_t)(m0 + row) * 1024 + k0 + c0;
            const float4* src = (const float4*)base;
            float4 f0 = src[0], f1 = src[1], f2 = src[2], f3 = src[3];
            *(uint4*)&Atile[row * 32 + c0]     = cvt8_bf16(f0, f1);
            *(uint4*)&Atile[row * 32 + c0 + 8] = cvt8_bf16(f2, f3);
            if (k0 + 32 < 1024) __builtin_prefetch(base + 32, 0, 1);
        }
        // stage B: 64 rows of W x 32 cols (4 threads per row)
        {
            int row = tid >> 2;
            int c0  = (tid & 3) * 8;
            const float* base = W + (size_t)(n0 + row) * 1024 + k0 + c0;
            const float4* src = (const float4*)base;
            float4 w0 = src[0], w1 = src[1];
            *(uint4*)&Btile[row * 32 + c0] = cvt8_bf16(w0, w1);
            if (k0 + 32 < 1024) __builtin_prefetch(base + 32, 0, 1);
        }
        __syncthreads();

        v16bf af = load_a_frag(&Atile[wave * 16 * 32], 32, lane);
#pragma unroll
        for (int j = 0; j < 4; ++j) {
            v16bf bfr = load_b_frag(&Btile[j * 16 * 32], 32, lane);
            acc[j] = wmma_bf16(af, bfr, acc[j]);
        }
        __syncthreads();
    }

    const int nlo = lane & 15;
    const int lh  = lane >> 4;
#pragma unroll
    for (int j = 0; j < 4; ++j) {
        int ng = n0 + j * 16 + nlo;
        float bval = bias[ng];
#pragma unroll
        for (int r = 0; r < 8; ++r) {
            int mg = m0 + wave * 16 + r + lh * 8;
            float val = acc[j][r] + bval;
            if (mode == 0) {
                ((float*)out)[(size_t)mg * 1024 + ng] = val;
            } else {
                int b = mg >> 11, s = mg & 2047;     // S = 2048
                int h = ng >> 6,  hd = ng & 63;      // HD = 64
                size_t idx = (mode == 1)
                    ? ((((size_t)b * 16 + h) * 2048 + s) * 64 + hd)    // [B,H,S,HD]
                    : ((((size_t)b * 16 + h) * 64 + hd) * 2048 + s);   // [B,H,HD,S]
                ((unsigned short*)out)[idx] = f2bf(val);
            }
        }
    }
}

// ---------------------------------------------------------------------------
// Flash attention: Q,K bf16 [B,H,S,HD]; V bf16 [B,H,HD,S] (pre-transposed).
// grid (S/128, H, B); 8 waves, wave owns 16 query rows. Key chunk 32,
// double-buffered LDS; K tile DMA'd by TDM on wave 0, V tile by TDM on
// wave 1 (both tracked by TENSORcnt). scores scaled by sqrt(HD)=8.
// ---------------------------------------------------------------------------
__global__ __launch_bounds__(256)
void attn_kernel(const unsigned short* __restrict__ Q,
                 const unsigned short* __restrict__ K,
                 const unsigned short* __restrict__ V,
                 float* __restrict__ Oattn)
{
    __shared__ __align__(16) unsigned short Ktile[2][32 * 64];   // [s_local][hd]
    __shared__ __align__(16) unsigned short Vtile[2][64 * 32];   // [hd][s_local]
    __shared__ __align__(16) unsigned short Ptile[8][16 * 32];

    const int tid  = threadIdx.x;
    const int lane = tid & 31;
    const int wave = tid >> 5;
    const int b    = blockIdx.z;
    const int h    = blockIdx.y;
    const int q0   = blockIdx.x * 128 + wave * 16;

    const size_t headoff = ((size_t)b * 16 + h) * 2048 * 64;  // same for both layouts
    const unsigned short* Qh = Q + headoff;
    const unsigned short* Kh = K + headoff;
    const unsigned short* Vh = V + headoff;                   // [HD=64][S=2048]

    const int m   = lane & 15;
    const int lh  = lane >> 4;
    const int nlo = lane & 15;

    // Q fragments once: hd chunks [0,32) and [32,64)
    FragBF qa0, qa1;
    {
        const unsigned short* qrow = Qh + (size_t)(q0 + m) * 64;
        qa0.q[0] = *(const uint4*)(qrow + lh * 8);
        qa0.q[1] = *(const uint4*)(qrow + 16 + lh * 8);
        qa1.q[0] = *(const uint4*)(qrow + 32 + lh * 8);
        qa1.q[1] = *(const uint4*)(qrow + 48 + lh * 8);
    }

    v8f o[4] = {};
    float mrow[8], lrow[8];
#pragma unroll
    for (int r = 0; r < 8; ++r) { mrow[r] = -1e30f; lrow[r] = 0.0f; }

    auto stage = [&](int kc, int buf) {
#ifdef HAVE_TDM
        // K: 32 rows x 64 elems, row stride 64  (issued by wave 0)
        if (wave == 0)
            tdm_load_2d(Kh + (size_t)kc * 64, lds_off_of(&Ktile[buf][0]), 64, 32, 64);
        // V: 64 rows x 32 elems, row stride 2048 (issued by wave 1)
        if (wave == 1)
            tdm_load_2d(Vh + kc, lds_off_of(&Vtile[buf][0]), 32, 64, 2048);
#else
        {
            int row = tid >> 3;                 // 32 rows, 8 threads/row
            int c0  = (tid & 7) * 8;
            *(uint4*)&Ktile[buf][row * 64 + c0] =
                *(const uint4*)(Kh + (size_t)(kc + row) * 64 + c0);
        }
        {
            int row = tid >> 2;                 // 64 rows, 4 threads/row
            int c0  = (tid & 3) * 8;
            *(uint4*)&Vtile[buf][row * 32 + c0] =
                *(const uint4*)(Vh + (size_t)row * 2048 + kc + c0);
        }
#endif
    };

    stage(0, 0);
    wait_tensorcnt0();
    __syncthreads();

    for (int kc = 0; kc < 2048; kc += 32) {
        const int buf = (kc >> 5) & 1;
        if (kc + 32 < 2048) stage(kc + 32, buf ^ 1);

        // scores: 16x32 tile = two 16x16 WMMAs per n-tile, K-contraction 64
        v8f s0 = {}, s1 = {};
        s0 = wmma_bf16(qa0.v, load_b_frag(&Ktile[buf][0 * 64 + 0],  64, lane), s0);
        s0 = wmma_bf16(qa1.v, load_b_frag(&Ktile[buf][0 * 64 + 32], 64, lane), s0);
        s1 = wmma_bf16(qa0.v, load_b_frag(&Ktile[buf][16 * 64 + 0],  64, lane), s1);
        s1 = wmma_bf16(qa1.v, load_b_frag(&Ktile[buf][16 * 64 + 32], 64, lane), s1);

        // online softmax; C layout: vgpr r holds row r (lanes 0-15) / r+8 (16-31)
        unsigned short* Pw = &Ptile[wave][0];
#pragma unroll
        for (int r = 0; r < 8; ++r) {
            float a = s0[r] * 8.0f;   // reference: scores / HD^-0.5
            float c = s1[r] * 8.0f;
            float mx = fmaxf(a, c);
            mx = fmaxf(mx, __shfl_xor(mx, 1, 32));
            mx = fmaxf(mx, __shfl_xor(mx, 2, 32));
            mx = fmaxf(mx, __shfl_xor(mx, 4, 32));
            mx = fmaxf(mx, __shfl_xor(mx, 8, 32));
            float mnew = fmaxf(mrow[r], mx);
            float corr = __expf(mrow[r] - mnew);
            float p0 = __expf(a - mnew);
            float p1 = __expf(c - mnew);
            float sum = p0 + p1;
            sum += __shfl_xor(sum, 1, 32);
            sum += __shfl_xor(sum, 2, 32);
            sum += __shfl_xor(sum, 4, 32);
            sum += __shfl_xor(sum, 8, 32);
            lrow[r] = lrow[r] * corr + sum;
            mrow[r] = mnew;
#pragma unroll
            for (int j = 0; j < 4; ++j) o[j][r] *= corr;

            int prow = r + lh * 8;
            Pw[prow * 32 + nlo]      = f2bf(p0);
            Pw[prow * 32 + 16 + nlo] = f2bf(p1);
        }
        // per-wave LDS round trip to re-stripe P into A-operand layout
        asm volatile("s_wait_dscnt 0" ::: "memory");

        FragBF pf;
        pf.q[0] = *(const uint4*)&Pw[m * 32 + lh * 8];
        pf.q[1] = *(const uint4*)&Pw[m * 32 + 16 + lh * 8];

#pragma unroll
        for (int j = 0; j < 4; ++j)
            o[j] = wmma_bf16(pf.v, load_b_frag(&Vtile[buf][j * 16 * 32], 32, lane), o[j]);

        wait_tensorcnt0();
        __syncthreads();
    }

    // normalize and write f32 [B,S,D]
#pragma unroll
    for (int r = 0; r < 8; ++r) {
        float inv = 1.0f / lrow[r];
        int qrow = q0 + r + lh * 8;
        float* orow = Oattn + ((size_t)(b * 2048 + qrow)) * 1024 + h * 64;
#pragma unroll
        for (int j = 0; j < 4; ++j)
            orow[j * 16 + nlo] = o[j][r] * inv;
    }
}

// ---------------------------------------------------------------------------
extern "C" void kernel_launch(void* const* d_in, const int* in_sizes, int n_in,
                              void* d_out, int out_size, void* d_ws, size_t ws_size,
                              hipStream_t stream) {
    const float* query = (const float*)d_in[0];
    const float* key   = (const float*)d_in[1];
    const float* value = (const float*)d_in[2];
    const float* Wq    = (const float*)d_in[3];
    const float* bq    = (const float*)d_in[4];
    const float* Wk    = (const float*)d_in[5];
    const float* bk    = (const float*)d_in[6];
    const float* Wv    = (const float*)d_in[7];
    const float* bv    = (const float*)d_in[8];
    const float* Wo    = (const float*)d_in[9];
    const float* bo    = (const float*)d_in[10];

    const size_t nElem = 4ull * 1024 * 1024;   // B*S*D = 4194304
    unsigned short* Qb = (unsigned short*)d_ws;
    unsigned short* Kb = Qb + nElem;
    unsigned short* Vb = Kb + nElem;           // transposed-per-head layout
    float* Xattn = (float*)(Vb + nElem);

    dim3 blk(256);
    dim3 gproj(16, 32);          // N/64, M/128 with M=4096, N=1024
    gemm_bias_kernel<<<gproj, blk, 0, stream>>>(query, Wq, bq, (void*)Qb, 1);
    gemm_bias_kernel<<<gproj, blk, 0, stream>>>(key,   Wk, bk, (void*)Kb, 1);
    gemm_bias_kernel<<<gproj, blk, 0, stream>>>(value, Wv, bv, (void*)Vb, 2);

    attn_kernel<<<dim3(16, 16, 2), blk, 0, stream>>>(Qb, Kb, Vb, Xattn);

    gemm_bias_kernel<<<gproj, blk, 0, stream>>>(Xattn, Wo, bo, d_out, 0);
}
